// MambaModel_37434934952492
// MI455X (gfx1250) — compile-verified, matrix operand order checked by reading, not run on hardware
//
#include <hip/hip_runtime.h>
#include <hip/hip_bf16.h>
#include <cstdint>
#include <cstddef>

// ---------------- model dims ----------------
#define VOCAB 50257
#define DMODEL 768
#define HDIM 128
#define NBLK 4
#define CONVK 4
#define BATCH 2
#define SEQ 1024
#define MROWS (BATCH * SEQ) // 2048
#define SCHUNK 64
#define NCHUNK (SEQ / SCHUNK) // 16

typedef __attribute__((ext_vector_type(16))) __bf16        v16bf;
typedef __attribute__((ext_vector_type(8)))  float         v8f;
typedef __attribute__((ext_vector_type(8)))  unsigned int  v8u;
typedef __attribute__((ext_vector_type(4)))  int           v4i;

#if __has_builtin(__builtin_amdgcn_global_load_async_to_lds_b128)
#define ASYNC_LDS 1
#else
#define ASYNC_LDS 0
#endif

// address-space casts for the async-to-LDS DMA builtin
#define GAS_V4I(p) ((__attribute__((address_space(1))) v4i*)(p))
#define LAS_V4I(p) ((__attribute__((address_space(3))) v4i*)(p))

static __device__ __forceinline__ unsigned short f32_to_bf16_bits(float f) {
  unsigned int u = __builtin_bit_cast(unsigned int, f);
  u += 0x7fffu + ((u >> 16) & 1u);          // round to nearest even
  return (unsigned short)(u >> 16);
}

static __device__ __forceinline__ void wait_async_zero() {
#if ASYNC_LDS
#if __has_builtin(__builtin_amdgcn_s_wait_asynccnt)
  __builtin_amdgcn_s_wait_asynccnt(0);
#else
  asm volatile("s_wait_asynccnt 0" ::: "memory");
#endif
#endif
}

// ---------------- embedding gather: h[row,:] = embed[x[row],:] ----------------
__global__ __launch_bounds__(256) void gather_embed_k(
    const int* __restrict__ x, const float* __restrict__ embed,
    float* __restrict__ h) {
  const int row = blockIdx.x;                     // 0..MROWS-1
  const int tok = x[row];
  const float* src = embed + (size_t)tok * DMODEL;
  float* dst = h + (size_t)row * DMODEL;
  for (int j = threadIdx.x; j < DMODEL; j += 256) dst[j] = src[j];
}

// ---------------- fp32 -> bf16 bulk convert ----------------
__global__ __launch_bounds__(256) void cvt_bf16_k(
    const float* __restrict__ src, unsigned short* __restrict__ dst, long n) {
  long i = (long)blockIdx.x * blockDim.x + threadIdx.x;
  const long stride = (long)gridDim.x * blockDim.x;
  for (; i < n; i += stride) dst[i] = f32_to_bf16_bits(src[i]);
}

// ---------------- transpose [R,C] f32 -> [C,R] bf16 ----------------
__global__ __launch_bounds__(256) void transpose_bf16_k(
    const float* __restrict__ src, unsigned short* __restrict__ dst,
    int R, int C) {
  long i = (long)blockIdx.x * blockDim.x + threadIdx.x;
  const long n = (long)R * C;
  const long stride = (long)gridDim.x * blockDim.x;
  for (; i < n; i += stride) {
    const int r = (int)(i / C), c = (int)(i % C);
    dst[(size_t)c * R + r] = f32_to_bf16_bits(src[i]);
  }
}

// ---------------- depthwise causal conv (K=4) + bias + SiLU -> bf16 ----------------
__global__ __launch_bounds__(256) void conv_silu_k(
    const float* __restrict__ h, const float* __restrict__ cw,
    const float* __restrict__ cb, unsigned short* __restrict__ u) {
  const int bl = blockIdx.x;                  // b*SEQ + l
  const int l = bl & (SEQ - 1);
  const float* base = h + (size_t)bl * DMODEL;
  for (int d = threadIdx.x; d < DMODEL; d += 256) {
    float acc = cb[d];
    #pragma unroll
    for (int j = 0; j < CONVK; ++j) {
      const int ll = l + j - (CONVK - 1);
      if (ll >= 0)
        acc += base[(long)(j - (CONVK - 1)) * DMODEL + d] * cw[d * CONVK + j];
    }
    const float s = acc / (1.f + __expf(-acc));   // SiLU
    u[(size_t)bl * DMODEL + d] = f32_to_bf16_bits(s);
  }
}

// ---------------- chunked associative SSM scan --------------------------------
// recurrence s <- d*s + i ; composition (d2,i2)o(d1,i1) = (d1*d2, d2*i1+i2)
// pass1: per-chunk elementwise decay/inc + chunk aggregates (depth SCHUNK)
__global__ __launch_bounds__(256) void scan_pass1_k(
    const float* __restrict__ dpre_bu,                    // [M, 2H] (dpre | Bu)
    const float* __restrict__ b_delta, const float* __restrict__ logA,
    float* __restrict__ decays, float* __restrict__ incs, // [M, H]
    float* __restrict__ chA, float* __restrict__ chB) {   // [B*NCHUNK, H]
  const int id = blockIdx.x * 256 + threadIdx.x;          // 0..4095
  const int hc = id & (HDIM - 1);
  const int t2 = id >> 7;                                 // 0..31
  const int c = t2 & (NCHUNK - 1);
  const int b = t2 >> 4;
  const float bd = b_delta[hc];
  const float Aneg = -__expf(logA[hc]);
  float Aagg = 1.f, Bagg = 0.f;
  const int row0 = b * SEQ + c * SCHUNK;
  for (int s = 0; s < SCHUNK; ++s) {
    const size_t r = (size_t)(row0 + s);
    const float dp = dpre_bu[r * (2 * HDIM) + hc] + bd;
    const float delta = (dp > 20.f) ? dp : log1pf(__expf(dp)); // softplus
    const float d = __expf(delta * Aneg);
    const float inc = delta * dpre_bu[r * (2 * HDIM) + HDIM + hc];
    decays[r * HDIM + hc] = d;
    incs[r * HDIM + hc] = inc;
    Aagg *= d;
    Bagg = d * Bagg + inc;
  }
  const size_t cidx = (size_t)(b * NCHUNK + c) * HDIM + hc;
  chA[cidx] = Aagg;
  chB[cidx] = Bagg;
}

// pass2: exclusive scan of chunk aggregates -> chunk entry states (depth NCHUNK)
__global__ __launch_bounds__(256) void scan_pass2_k(
    const float* __restrict__ chA, const float* __restrict__ chB,
    float* __restrict__ entry) {                          // [B*NCHUNK, H]
  const int t = threadIdx.x;                              // 256 == B*H
  const int b = t >> 7, hc = t & (HDIM - 1);
  float carry = 0.f;
  for (int c = 0; c < NCHUNK; ++c) {
    const size_t idx = (size_t)(b * NCHUNK + c) * HDIM + hc;
    entry[idx] = carry;
    carry = chA[idx] * carry + chB[idx];
  }
}

// pass3: replay within chunk from entry state, emit bf16 (depth SCHUNK)
__global__ __launch_bounds__(256) void scan_pass3_k(
    const float* __restrict__ decays, const float* __restrict__ incs,
    const float* __restrict__ entry, unsigned short* __restrict__ hs) {
  const int id = blockIdx.x * 256 + threadIdx.x;
  const int hc = id & (HDIM - 1);
  const int t2 = id >> 7;
  const int c = t2 & (NCHUNK - 1);
  const int b = t2 >> 4;
  float s = entry[(size_t)(b * NCHUNK + c) * HDIM + hc];
  const int row0 = b * SEQ + c * SCHUNK;
  for (int st = 0; st < SCHUNK; ++st) {
    const size_t r = (size_t)(row0 + st);
    s = decays[r * HDIM + hc] * s + incs[r * HDIM + hc];
    hs[r * HDIM + hc] = f32_to_bf16_bits(s);
  }
}

// ---------------- LayerNorm over D, emit bf16 ----------------
__global__ __launch_bounds__(256) void layernorm_bf16_k(
    const float* __restrict__ h, const float* __restrict__ gamma,
    const float* __restrict__ beta, unsigned short* __restrict__ hn) {
  __shared__ float s1[256], s2[256];
  const int row = blockIdx.x;
  const int tid = threadIdx.x;
  const float* src = h + (size_t)row * DMODEL;
  float v[3], a = 0.f, b = 0.f;
  #pragma unroll
  for (int j = 0; j < 3; ++j) {
    v[j] = src[tid + 256 * j];
    a += v[j];
    b += v[j] * v[j];
  }
  s1[tid] = a; s2[tid] = b;
  __syncthreads();
  for (int off = 128; off > 0; off >>= 1) {
    if (tid < off) { s1[tid] += s1[tid + off]; s2[tid] += s2[tid + off]; }
    __syncthreads();
  }
  const float mean = s1[0] * (1.f / DMODEL);
  const float var  = s2[0] * (1.f / DMODEL) - mean * mean;
  const float inv  = rsqrtf(var + 1e-5f);
  #pragma unroll
  for (int j = 0; j < 3; ++j) {
    const int c = tid + 256 * j;
    const float nv = (v[j] - mean) * inv * gamma[c] + beta[c];
    hn[(size_t)row * DMODEL + c] = f32_to_bf16_bits(nv);
  }
}

// ---------------- WMMA bf16 GEMM: C[M,N] = A[M,K] * Bt[N,K]^T (+resid) -------
// 256 threads = 8 wave32; 128x128 tile, K-step 32, wave sub-tile 32x64.
#define LDSS 40   // LDS row stride in ushorts: 80B, 16B-aligned rows

static __device__ __forceinline__ void gemm_compute_step(
    const unsigned short* __restrict__ As, const unsigned short* __restrict__ Bs,
    int mbase, int nbase, int lrow, const int* koff, v8f acc[2][4]) {
  v16bf afrag[2];
  #pragma unroll
  for (int im = 0; im < 2; ++im) {
    const unsigned short* r = As + (mbase + im * 16 + lrow) * LDSS;
    v8u bits;
    #pragma unroll
    for (int j = 0; j < 8; ++j)
      bits[j] = *(const unsigned int*)(r + koff[j]);
    afrag[im] = __builtin_bit_cast(v16bf, bits);
  }
  #pragma unroll
  for (int jn = 0; jn < 4; ++jn) {
    const unsigned short* r = Bs + (nbase + jn * 16 + lrow) * LDSS;
    v8u bits;
    #pragma unroll
    for (int j = 0; j < 8; ++j)
      bits[j] = *(const unsigned int*)(r + koff[j]);
    const v16bf bfrag = __builtin_bit_cast(v16bf, bits);
    acc[0][jn] = __builtin_amdgcn_wmma_f32_16x16x32_bf16(
        false, afrag[0], false, bfrag, (short)0, acc[0][jn], false, false);
    acc[1][jn] = __builtin_amdgcn_wmma_f32_16x16x32_bf16(
        false, afrag[1], false, bfrag, (short)0, acc[1][jn], false, false);
  }
}

__global__ __launch_bounds__(256) void wmma_gemm_bf16_k(
    const unsigned short* __restrict__ A,   // [M,K] bf16 bits
    const unsigned short* __restrict__ Bt,  // [N,K] bf16 bits (B transposed)
    float* __restrict__ C,                  // [M,N] f32
    const float* __restrict__ resid,        // [M,N] f32 or nullptr
    int Mdim, int Ndim, int Kdim) {
  const int tid = threadIdx.x;
  const int n0 = blockIdx.x * 128;
  const int m0 = blockIdx.y * 128;

  const int wave = tid >> 5, lane = tid & 31;
  const int wm = wave & 3, wn = wave >> 2;  // 4x2 wave grid
  const int mbase = wm * 32, nbase = wn * 64;
  const int lrow = lane & 15, lgrp = lane >> 4;

  // ISA 7.12.2 16-bit A-matrix K mapping per lane group / vgpr pair
  int koff[8];
  #pragma unroll
  for (int j = 0; j < 8; ++j)
    koff[j] = ((j < 4) ? 0 : 16) + 8 * lgrp + 2 * (j & 3);

  v8f acc[2][4];
  #pragma unroll
  for (int im = 0; im < 2; ++im)
    #pragma unroll
    for (int jn = 0; jn < 4; ++jn)
      acc[im][jn] = (v8f){0.f, 0.f, 0.f, 0.f, 0.f, 0.f, 0.f, 0.f};

  // cooperative tile loaders: 2 threads per row, 32 bytes each
  const int ldrow = tid >> 1;
  const int ldhalf = (tid & 1) * 16;        // ushort offset
  const int arow = m0 + ldrow;              // M always a multiple of 128
  const int braw = n0 + ldrow;
  const int brow = (braw < Ndim) ? braw : (Ndim - 1); // clamp vocab edge
  const int ksteps = Kdim >> 5;

#if ASYNC_LDS
  // double-buffered DMA pipeline: GLOBAL_LOAD_ASYNC_TO_LDS (ASYNCcnt) + 1 barrier
  __shared__ unsigned short As[2][128 * LDSS];
  __shared__ unsigned short Bs[2][128 * LDSS];
  auto issue = [&](int buf, int kk) {
    const unsigned short* ga = A + (size_t)arow * Kdim + kk + ldhalf;
    const unsigned short* gb = Bt + (size_t)brow * Kdim + kk + ldhalf;
    unsigned short* la = &As[buf][ldrow * LDSS + ldhalf];
    unsigned short* lb = &Bs[buf][ldrow * LDSS + ldhalf];
    __builtin_amdgcn_global_load_async_to_lds_b128(
        GAS_V4I(ga), LAS_V4I(la), 0, 0);
    __builtin_amdgcn_global_load_async_to_lds_b128(
        GAS_V4I(ga + 8), LAS_V4I(la + 8), 0, 0);
    __builtin_amdgcn_global_load_async_to_lds_b128(
        GAS_V4I(gb), LAS_V4I(lb), 0, 0);
    __builtin_amdgcn_global_load_async_to_lds_b128(
        GAS_V4I(gb + 8), LAS_V4I(lb + 8), 0, 0);
  };
  issue(0, 0);
  for (int ks = 0; ks < ksteps; ++ks) {
    const int cur = ks & 1;
    wait_async_zero();      // my tile[cur] DMA landed in LDS
    __syncthreads();        // everyone's landed; tile[cur^1] readers done
    if (ks + 1 < ksteps) issue(cur ^ 1, (ks + 1) << 5); // overlap with WMMA
    gemm_compute_step(As[cur], Bs[cur], mbase, nbase, lrow, koff, acc);
  }
#else
  // fallback: VGPR-staged single buffer (compiler software-pipelines loads)
  __shared__ unsigned short As[128 * LDSS];
  __shared__ unsigned short Bs[128 * LDSS];
  for (int ks = 0; ks < ksteps; ++ks) {
    const int kk = ks << 5;
    const uint4* ap = (const uint4*)(A + (size_t)arow * Kdim + kk + ldhalf);
    const uint4* bp = (const uint4*)(Bt + (size_t)brow * Kdim + kk + ldhalf);
    const uint4 av0 = ap[0], av1 = ap[1];
    const uint4 bv0 = bp[0], bv1 = bp[1];
    if (ks + 1 < ksteps) {                  // global_prefetch_b8 of next K-slab
      __builtin_prefetch((const void*)(ap + 4), 0, 0);
      __builtin_prefetch((const void*)(bp + 4), 0, 0);
    }
    __syncthreads();                        // previous compute done
    unsigned int* asl = (unsigned int*)(As + ldrow * LDSS + ldhalf);
    unsigned int* bsl = (unsigned int*)(Bs + ldrow * LDSS + ldhalf);
    asl[0] = av0.x; asl[1] = av0.y; asl[2] = av0.z; asl[3] = av0.w;
    asl[4] = av1.x; asl[5] = av1.y; asl[6] = av1.z; asl[7] = av1.w;
    bsl[0] = bv0.x; bsl[1] = bv0.y; bsl[2] = bv0.z; bsl[3] = bv0.w;
    bsl[4] = bv1.x; bsl[5] = bv1.y; bsl[6] = bv1.z; bsl[7] = bv1.w;
    __syncthreads();                        // tile visible
    gemm_compute_step(As, Bs, mbase, nbase, lrow, koff, acc);
  }
#endif

  // D layout (ISA 7.12.2): vgpr r, lane = n + 16*(m/8); m = r + 8*lgrp
  #pragma unroll
  for (int im = 0; im < 2; ++im) {
    #pragma unroll
    for (int jn = 0; jn < 4; ++jn) {
      const int gn = n0 + nbase + jn * 16 + lrow;
      if (gn < Ndim) {
        const int mrow0 = m0 + mbase + im * 16 + lgrp * 8;
        #pragma unroll
        for (int rr = 0; rr < 8; ++rr) {
          const size_t idx = (size_t)(mrow0 + rr) * Ndim + gn;
          float v = acc[im][jn][rr];
          if (resid) v += resid[idx];
          C[idx] = v;
        }
      }
    }
  }
}

// ---------------- host-side orchestration ----------------
extern "C" void kernel_launch(void* const* d_in, const int* in_sizes, int n_in,
                              void* d_out, int out_size, void* d_ws, size_t ws_size,
                              hipStream_t stream) {
  const int*   x       = (const int*)d_in[0];
  const float* embed   = (const float*)d_in[1];
  const float* conv_w  = (const float*)d_in[2];
  const float* conv_b  = (const float*)d_in[3];
  const float* W_delta = (const float*)d_in[4];
  const float* b_delta = (const float*)d_in[5];
  const float* W_in    = (const float*)d_in[6];
  const float* W_out   = (const float*)d_in[7];
  const float* log_A   = (const float*)d_in[8];
  const float* gamma   = (const float*)d_in[9];
  const float* beta    = (const float*)d_in[10];
  float* out = (float*)d_out;

  char* wp = (char*)d_ws;
  auto carve = [&](size_t bytes) -> char* {
    char* p = wp;
    wp += (bytes + 255) & ~(size_t)255;
    return p;
  };
  unsigned short* embed_bf = (unsigned short*)carve((size_t)VOCAB * DMODEL * 2);
  float*          hbuf     = (float*)carve((size_t)MROWS * DMODEL * 4);
  unsigned short* u_bf     = (unsigned short*)carve((size_t)MROWS * DMODEL * 2);
  unsigned short* wdin_t   = (unsigned short*)carve((size_t)2 * HDIM * DMODEL * 2);
  unsigned short* wout_t   = (unsigned short*)carve((size_t)DMODEL * HDIM * 2);
  float*          dpre_bu  = (float*)carve((size_t)MROWS * 2 * HDIM * 4);
  float*          decays   = (float*)carve((size_t)MROWS * HDIM * 4);
  float*          incs     = (float*)carve((size_t)MROWS * HDIM * 4);
  float*          chA      = (float*)carve((size_t)BATCH * NCHUNK * HDIM * 4);
  float*          chB      = (float*)carve((size_t)BATCH * NCHUNK * HDIM * 4);
  float*          entry    = (float*)carve((size_t)BATCH * NCHUNK * HDIM * 4);
  unsigned short* hs_bf    = (unsigned short*)carve((size_t)MROWS * HDIM * 2);
  unsigned short* hn_bf    = (unsigned short*)carve((size_t)MROWS * DMODEL * 2);

  // embed -> bf16 (head B matrix, already [N=V, K=D])
  cvt_bf16_k<<<4096, 256, 0, stream>>>(embed, embed_bf, (long)VOCAB * DMODEL);
  // embedding gather (fp32 residual stream)
  gather_embed_k<<<MROWS, 256, 0, stream>>>(x, embed, hbuf);

  for (int i = 0; i < NBLK; ++i) {
    conv_silu_k<<<MROWS, 256, 0, stream>>>(
        hbuf, conv_w + (size_t)i * DMODEL * CONVK, conv_b + (size_t)i * DMODEL, u_bf);

    // fused [Wd | Win] -> [2H, D] bf16
    transpose_bf16_k<<<512, 256, 0, stream>>>(
        W_delta + (size_t)i * DMODEL * HDIM, wdin_t, DMODEL, HDIM);
    transpose_bf16_k<<<512, 256, 0, stream>>>(
        W_in + (size_t)i * DMODEL * HDIM, wdin_t + (size_t)HDIM * DMODEL,
        DMODEL, HDIM);
    transpose_bf16_k<<<512, 256, 0, stream>>>(
        W_out + (size_t)i * HDIM * DMODEL, wout_t, HDIM, DMODEL);

    // [delta_pre | Bu] = u @ [Wd | Win]   (M=2048, N=256, K=768)
    {
      dim3 grid(2 * HDIM / 128, MROWS / 128);
      wmma_gemm_bf16_k<<<grid, 256, 0, stream>>>(u_bf, wdin_t, dpre_bu, nullptr,
                                                 MROWS, 2 * HDIM, DMODEL);
    }

    // chunked scan: depth 1024 -> 64 + 16 + 64
    scan_pass1_k<<<16, 256, 0, stream>>>(dpre_bu, b_delta + (size_t)i * HDIM,
                                         log_A + (size_t)i * HDIM,
                                         decays, incs, chA, chB);
    scan_pass2_k<<<1, 256, 0, stream>>>(chA, chB, entry);
    scan_pass3_k<<<16, 256, 0, stream>>>(decays, incs, entry, hs_bf);

    // h += hs @ Wout   (M=2048, N=768, K=128), in-place residual
    {
      dim3 grid(DMODEL / 128, MROWS / 128);
      wmma_gemm_bf16_k<<<grid, 256, 0, stream>>>(hs_bf, wout_t, hbuf, hbuf,
                                                 MROWS, DMODEL, HDIM);
    }
  }

  layernorm_bf16_k<<<MROWS, 256, 0, stream>>>(hbuf, gamma, beta, hn_bf);

  // head: out = hn @ embed^T   (M=2048, N=50257, K=768)
  {
    dim3 grid((VOCAB + 127) / 128, MROWS / 128);
    wmma_gemm_bf16_k<<<grid, 256, 0, stream>>>(hn_bf, embed_bf, out, nullptr,
                                               MROWS, VOCAB, DMODEL);
  }
}
